// VectorQuantize_8211977470202
// MI455X (gfx1250) — compile-verified, matrix operand order checked by reading, not run on hardware
//
#include <hip/hip_runtime.h>
#include <hip/hip_bf16.h>

// Problem constants (fixed by the reference)
#define N_PTS  32768
#define DIM    256
#define NCLUS  4096
#define NTILES (NCLUS / 16)
#define DECAY  0.8f
#define ONE_M_DECAY 0.2f
#define EPS_F  1e-05f

typedef __attribute__((ext_vector_type(16))) __bf16 v16bf;
typedef __attribute__((ext_vector_type(8)))  float  v8f;
typedef __attribute__((ext_vector_type(4))) unsigned int u32x4;
typedef __attribute__((ext_vector_type(8))) int          i32x8;
typedef __attribute__((ext_vector_type(4))) int          i32x4;

union Frag { v16bf v; uint4 q[2]; };

#if defined(__AMDGCN__) && __has_builtin(__builtin_amdgcn_tensor_load_to_lds) && \
    __has_builtin(__builtin_amdgcn_s_wait_tensorcnt)
#define VQ_USE_TDM 1
#else
#define VQ_USE_TDM 0
#endif

// ---------------------------------------------------------------------------
// Prep kernels
// ---------------------------------------------------------------------------
__global__ __launch_bounds__(256) void vq_zero(float* p, int n) {
    int i = blockIdx.x * 256 + threadIdx.x;
    if (i < n) p[i] = 0.0f;
}

__global__ __launch_bounds__(256) void vq_cvt_x(const float* __restrict__ x,
                                                __bf16* __restrict__ xbf, int n) {
    int i = blockIdx.x * 256 + threadIdx.x;
    if (i < n) xbf[i] = (__bf16)x[i];
}

// One block (256 threads) per codebook row: bf16 convert + e2[c] = sum(e*e)
__global__ __launch_bounds__(256) void vq_prep_embed(const float* __restrict__ embed,
                                                     __bf16* __restrict__ ebf,
                                                     float* __restrict__ e2) {
    __shared__ float red[256];
    int c = blockIdx.x;
    int d = threadIdx.x;
    float v = embed[(size_t)c * DIM + d];
    ebf[(size_t)c * DIM + d] = (__bf16)v;
    red[d] = v * v;
    __syncthreads();
    for (int s = 128; s > 0; s >>= 1) {
        if (d < s) red[d] += red[d + s];
        __syncthreads();
    }
    if (d == 0) e2[c] = red[0];
}

// ---------------------------------------------------------------------------
// TDM helper: DMA a contiguous run of n_elems bf16 (one "row") into LDS.
// D# packed per CDNA5 ISA ch.8: group0 = {flags, lds_addr, global_addr, type=2},
// group1 = {data_size=2B, tensor_dim0=n, tensor_dim1=1, tile_dim0=n, tile_dim1=1}.
// ---------------------------------------------------------------------------
#if VQ_USE_TDM
__device__ __forceinline__ void tdm_load_row(const __bf16* gsrc,
                                             unsigned lds_byte_addr, int n_elems) {
    unsigned long long ga = (unsigned long long)(uintptr_t)gsrc;
    u32x4 g0;
    g0[0] = 1u;                                      // count=1 (valid user D#)
    g0[1] = lds_byte_addr;                           // lds_addr [63:32]
    g0[2] = (unsigned)ga;                            // global_addr [95:64]
    g0[3] = (unsigned)(ga >> 32) | 0x80000000u;      // global_addr[56:32] | type=2
    i32x8 g1;
    g1[0] = 0x00010000;                              // data_size=1 (2 bytes/elem)
    g1[1] = (int)(((unsigned)n_elems & 0xffffu) << 16);        // tensor_dim0 lo16
    g1[2] = (int)(((unsigned)n_elems >> 16) | (1u << 16));     // dim0 hi | tensor_dim1=1
    g1[3] = (int)((unsigned)n_elems << 16);          // tile_dim0 [127:112]
    g1[4] = 1;                                       // tile_dim1=1, tile_dim2 unused
    g1[5] = n_elems;                                 // tensor_dim0_stride (1 row)
    g1[6] = 0;
    g1[7] = 0;
    i32x4 gz = {0, 0, 0, 0};
#if __clang_major__ >= 23
    i32x8 gz8 = {0, 0, 0, 0, 0, 0, 0, 0};
    __builtin_amdgcn_tensor_load_to_lds(g0, g1, gz, gz, gz8, 0);
#else
    __builtin_amdgcn_tensor_load_to_lds(g0, g1, gz, gz, 0);
#endif
}
#endif

// ---------------------------------------------------------------------------
// Fused distance-GEMM + argmax.
//   128 threads = 4 waves; wave w owns M-tile (blockIdx.x*4 + w), 16 rows,
//   A fragments resident in VGPRs. Per 16-column tile: 8x
//   v_wmma_f32_16x16x32_bf16 over K=256, epilogue argmax of (2*xe - e2).
//   Embed tiles are double-buffered in LDS; staged by the Tensor Data Mover
//   (TENSORcnt) when available, else a register-staged copy. One barrier/iter;
//   tile t+1 DMA overlaps tile t WMMAs.
// ---------------------------------------------------------------------------
__global__ __launch_bounds__(128) void vq_gemm_argmax(
        const __bf16* __restrict__ xbf, const __bf16* __restrict__ ebf,
        const float* __restrict__ e2, int* __restrict__ ind_i,
        float* __restrict__ ind_f) {
    __shared__ __align__(16) __bf16 sB[2][16 * DIM];   // 2 x 8KB column tiles

    const int lane = threadIdx.x & 31;
    const int wave = threadIdx.x >> 5;
    const int mtile = blockIdx.x * 4 + wave;
    const int hi = lane >> 4;               // which K half this lane holds
    const int row = (mtile << 4) + (lane & 15);

    // Preload A fragments: 8 chunks of K=32 bf16 (two 16B loads each)
    Frag A[8];
    const uint4* xrow = (const uint4*)(xbf + (size_t)row * DIM);
#pragma unroll
    for (int k = 0; k < 8; ++k) {
        A[k].q[0] = xrow[k * 4 + hi];       // K = k*32 + hi*8 .. +7
        A[k].q[1] = xrow[k * 4 + hi + 2];   // K = k*32 + 16 + hi*8 .. +7
    }

    float best[8];
    int   bidx[8];
#pragma unroll
    for (int r = 0; r < 8; ++r) { best[r] = -3.4e38f; bidx[r] = 0; }

#if VQ_USE_TDM
    // Each wave DMAs its own 4-row (2KB) slice of the tile via the TDM.
    const int uw = __builtin_amdgcn_readfirstlane(wave);
    const unsigned ldsb[2] = { (unsigned)(uintptr_t)&sB[0][0],
                               (unsigned)(uintptr_t)&sB[1][0] };
    // prologue: tile 0 -> buffer 0
    tdm_load_row(ebf + (size_t)uw * 4 * DIM, ldsb[0] + (unsigned)uw * 4 * DIM * 2, 4 * DIM);
#else
    // prologue: register-staged copy of tile 0 -> buffer 0
    uint4 st[4];
    {
        const uint4* s0 = (const uint4*)ebf;
#pragma unroll
        for (int i = 0; i < 4; ++i) st[i] = s0[threadIdx.x + i * 128];
        uint4* d0 = (uint4*)&sB[0][0];
#pragma unroll
        for (int i = 0; i < 4; ++i) d0[threadIdx.x + i * 128] = st[i];
    }
#endif

    for (int ct = 0; ct < NTILES; ++ct) {
#if VQ_USE_TDM
        __builtin_amdgcn_s_wait_tensorcnt(0);   // my slice of tile ct is in LDS
        __syncthreads();                        // all slices in; buf^1 readers done
        if (ct + 1 < NTILES)                    // DMA tile ct+1, overlaps WMMAs below
            tdm_load_row(ebf + ((size_t)(ct + 1) * 16 + (size_t)uw * 4) * DIM,
                         ldsb[(ct + 1) & 1] + (unsigned)uw * 4 * DIM * 2, 4 * DIM);
#else
        __syncthreads();                        // tile ct visible; buf^1 readers done
        if (ct + 1 < NTILES) {                  // issue global loads early
            const uint4* s = (const uint4*)(ebf + (size_t)(ct + 1) * 16 * DIM);
#pragma unroll
            for (int i = 0; i < 4; ++i) st[i] = s[threadIdx.x + i * 128];
            if (ct + 2 < NTILES)
                __builtin_prefetch(ebf + (size_t)(ct + 2) * 16 * DIM + (threadIdx.x << 5), 0, 0);
        }
#endif

        v8f acc = {};
        const __bf16* bbase = &sB[ct & 1][0];
        const uint4* brow = (const uint4*)(bbase + (size_t)(lane & 15) * DIM);
#pragma unroll
        for (int k = 0; k < 8; ++k) {
            Frag B;
            B.q[0] = brow[k * 4 + hi];
            B.q[1] = brow[k * 4 + hi + 2];
            acc = __builtin_amdgcn_wmma_f32_16x16x32_bf16(
                false, A[k].v, false, B.v, (short)0, acc, false, false);
        }

        // Epilogue: dist ~ 2*xe - e2 ; running argmax per (row, lane-column)
        const float e2l = e2[(ct << 4) + (lane & 15)];
        const int col = (ct << 4) + (lane & 15);
#pragma unroll
        for (int r = 0; r < 8; ++r) {
            float cand = 2.0f * acc[r] - e2l;
            if (cand > best[r]) { best[r] = cand; bidx[r] = col; }
        }

#if !VQ_USE_TDM
        if (ct + 1 < NTILES) {                  // publish tile ct+1 to the other buffer
            uint4* d = (uint4*)&sB[(ct + 1) & 1][0];
#pragma unroll
            for (int i = 0; i < 4; ++i) d[threadIdx.x + i * 128] = st[i];
        }
#endif
    }

    // Cross-lane argmax across the 16 column-lanes of each half-wave.
    // C/D layout: lanes 0-15 hold rows M=0..7 (vgpr r -> M=r), lanes 16-31 M=8..15.
#pragma unroll
    for (int m = 1; m < 16; m <<= 1) {
#pragma unroll
        for (int r = 0; r < 8; ++r) {
            float ov = __shfl_xor(best[r], m, 32);
            int   oi = __shfl_xor(bidx[r], m, 32);
            if (ov > best[r] || (ov == best[r] && oi < bidx[r])) {
                best[r] = ov; bidx[r] = oi;
            }
        }
    }
    if ((lane & 15) == 0) {
        int base = (mtile << 4) + (hi << 3);
#pragma unroll
        for (int r = 0; r < 8; ++r) {
            ind_i[base + r] = bidx[r];
            ind_f[base + r] = (float)bidx[r];
        }
    }
}

// ---------------------------------------------------------------------------
// Gather quantize, scatter segment sums (float atomics), commit-loss partials
// ---------------------------------------------------------------------------
__global__ __launch_bounds__(256) void vq_scatter(
        const float* __restrict__ x, const float* __restrict__ embed,
        const int* __restrict__ ind, float* __restrict__ out_q,
        float* __restrict__ counts, float* __restrict__ embed_sum,
        float* __restrict__ loss) {
    __shared__ float red[256];
    int n = blockIdx.x;
    int d = threadIdx.x;
    int c = ind[n];
    float xv = x[(size_t)n * DIM + d];
    float qv = embed[(size_t)c * DIM + d];
    out_q[(size_t)n * DIM + d] = qv;                 // quantize_st == quantize
    atomicAdd(&embed_sum[(size_t)c * DIM + d], xv);  // segment_sum(x)
    float diff = qv - xv;
    red[d] = diff * diff;
    __syncthreads();
    for (int s = 128; s > 0; s >>= 1) {
        if (d < s) red[d] += red[d + s];
        __syncthreads();
    }
    if (d == 0) {
        atomicAdd(loss, red[0]);
        atomicAdd(&counts[c], 1.0f);
    }
}

// new_cluster_size + total reduction + commit-loss finalize
__global__ __launch_bounds__(256) void vq_cluster(
        const float* __restrict__ cs, const float* __restrict__ counts,
        float* __restrict__ out_ncs, float* __restrict__ total,
        const float* __restrict__ loss, float* __restrict__ out_loss) {
    __shared__ float red[256];
    int c = blockIdx.x * 256 + threadIdx.x;
    float v = cs[c] * DECAY + counts[c] * ONE_M_DECAY;
    out_ncs[c] = v;
    red[threadIdx.x] = v;
    __syncthreads();
    for (int s = 128; s > 0; s >>= 1) {
        if (threadIdx.x < s) red[threadIdx.x] += red[threadIdx.x + s];
        __syncthreads();
    }
    if (threadIdx.x == 0) atomicAdd(total, red[0]);
    if (c == 0) out_loss[0] = loss[0] * (1.0f / ((float)N_PTS * (float)DIM));
}

// new_embed = EMA(embed_avg) / laplace-smoothed cluster size
__global__ __launch_bounds__(256) void vq_embed_update(
        const float* __restrict__ embed_avg, const float* __restrict__ embed_sum,
        const float* __restrict__ ncs, const float* __restrict__ total,
        float* __restrict__ out_embed) {
    int i = blockIdx.x * 256 + threadIdx.x;   // over C*D
    int c = i >> 8;
    float t = *total;
    float s = (ncs[c] + EPS_F) / (t + (float)NCLUS * EPS_F) * t;
    out_embed[i] = (embed_avg[i] * DECAY + embed_sum[i] * ONE_M_DECAY) / s;
}

// ---------------------------------------------------------------------------
extern "C" void kernel_launch(void* const* d_in, const int* in_sizes, int n_in,
                              void* d_out, int out_size, void* d_ws, size_t ws_size,
                              hipStream_t stream) {
    const float* x         = (const float*)d_in[0];   // (N, D)
    const float* embed     = (const float*)d_in[1];   // (1, C, D)
    const float* cs        = (const float*)d_in[2];   // (1, C)
    const float* embed_avg = (const float*)d_in[3];   // (1, C, D)

    // Output layout (floats, reference return order)
    float* out       = (float*)d_out;
    float* out_q     = out;                                   // N*D
    float* out_ind   = out + (size_t)N_PTS * DIM;             // N
    float* out_loss  = out_ind + N_PTS;                       // 1
    float* out_ne    = out_loss + 1;                          // C*D
    float* out_ncs   = out_ne + (size_t)NCLUS * DIM;          // C

    // Workspace layout (~22.2 MB)
    char* ws = (char*)d_ws;
    __bf16* xbf       = (__bf16*)(ws);                                  // 16 MB
    __bf16* ebf       = (__bf16*)(ws + 16777216);                       // 2 MB
    float*  e2        = (float*)(ws + 18874368);                        // 16 KB
    float*  counts    = (float*)(ws + 18890752);                        // 16 KB
    float*  embed_sum = (float*)(ws + 18907136);                        // 4 MB
    float*  loss_acc  = (float*)(ws + 23101440);                        // 4 B
    float*  total     = (float*)(ws + 23101444);                        // 4 B
    int*    ind       = (int*)  (ws + 23101448);                        // 128 KB

    // 1. zero accumulators (counts + embed_sum + loss + total, contiguous)
    {
        int nz = NCLUS + NCLUS * DIM + 2;
        vq_zero<<<(nz + 255) / 256, 256, 0, stream>>>(counts, nz);
    }
    // 2. precision prep
    vq_cvt_x<<<(N_PTS * DIM) / 256, 256, 0, stream>>>(x, xbf, N_PTS * DIM);
    vq_prep_embed<<<NCLUS, 256, 0, stream>>>(embed, ebf, e2);
    // 3. fused distance GEMM + argmax (WMMA bf16 -> f32, TDM-staged tiles)
    vq_gemm_argmax<<<N_PTS / 64, 128, 0, stream>>>(xbf, ebf, e2, ind, out_ind);
    // 4. gather quantize + scatter segment sums + loss
    vq_scatter<<<N_PTS, 256, 0, stream>>>(x, embed, ind, out_q, counts, embed_sum, loss_acc);
    // 5. EMA cluster sizes + total + loss finalize
    vq_cluster<<<NCLUS / 256, 256, 0, stream>>>(cs, counts, out_ncs, total, loss_acc, out_loss);
    // 6. new codebook
    vq_embed_update<<<(NCLUS * DIM) / 256, 256, 0, stream>>>(embed_avg, embed_sum, out_ncs, total, out_ne);
}
